// LSTMDecoder_6803228196931
// MI455X (gfx1250) — matrix operand order repaired and checked by reading, not compiled, not run on hardware
//
#include <hip/hip_runtime.h>

// ---------------------------------------------------------------------------
// LSTM decoder for MI455X (gfx1250, wave32, WMMA).
//   V=32000, E=H=512, B=16, T=256, 4H=2048, B*T=4096
// Plan:
//   1) convert weights f32->f16 (ws), bias sums
//   2) embedding gather -> x_f16 [T*16, 512]   (row = t*16 + b)
//   3) WMMA GEMM: preact0 = x @ Wi0^T + (bi0+bh0)   [4096, 2048] f32
//   4) sequential LSTM: 32 WGs (1 per WGP), weight slices resident in LDS
//      (195KB/WGP), h0/h1 double-buffered in global, L2-atomic barrier
//   5) WMMA GEMM: out = feats @ Wout^T + bout  [4096, 32000] f32
// GEMMs: 32x64 tile per wave, ping-pong double-buffered fragments (unroll-2
// k-loop, disjoint register sets) -> partial s_wait_loadcnt, no rotation movs.
// ---------------------------------------------------------------------------

typedef __attribute__((ext_vector_type(16))) _Float16 v16h;
typedef __attribute__((ext_vector_type(8)))  _Float16 v8h;
typedef __attribute__((ext_vector_type(8)))  float    v8f;

#define WMMA16(a, b, c) \
  __builtin_amdgcn_wmma_f32_16x16x32_f16(false, (a), false, (b), (short)0, (c), false, false)

__device__ __forceinline__ v8f vzero8() {
  v8f z;
#pragma unroll
  for (int i = 0; i < 8; ++i) z[i] = 0.0f;
  return z;
}

// Combine two 8xf16 chunks into one 16xf16 fragment register set.
__device__ __forceinline__ v16h combine16(const _Float16* p0, const _Float16* p1) {
  v8h a = *(const v8h*)p0;
  v8h b = *(const v8h*)p1;
  v16h o;
#pragma unroll
  for (int i = 0; i < 8; ++i) { o[i] = a[i]; o[i + 8] = b[i]; }
  return o;
}

// A fragment (16x32, row-major [16, lda] f16 tile), per ISA 16-bit A layout:
// lanes 0-15 row r: K = kb+0..7 and kb+16..23 ; lanes 16-31: +8 offset.
__device__ __forceinline__ v16h load_a_frag(const _Float16* tile, int lda, int kb,
                                            int r, int hi) {
  const _Float16* p = tile + (long)r * lda + kb + hi * 8;
  return combine16(p, p + 16);
}

// B fragment (32x16) from weights stored [N, K] row-major (i.e. W for x@W^T):
// lane covers output col n = row of W, K = kb + hi*8 + 0..7 and kb + 16 + hi*8 + 0..7.
__device__ __forceinline__ v16h load_b_frag(const _Float16* w_row_base, int kb, int hi) {
  const _Float16* p = w_row_base + kb + hi * 8;
  return combine16(p, p + 16);
}

__device__ __forceinline__ float sigf(float x) { return 1.0f / (1.0f + __expf(-x)); }

// ------------------------------- utility kernels ---------------------------

__global__ void cvt_f16(const float* __restrict__ s, _Float16* __restrict__ d, long n) {
  long i = (long)blockIdx.x * blockDim.x + threadIdx.x;
  long st = (long)gridDim.x * blockDim.x;
  for (; i < n; i += st) d[i] = (_Float16)s[i];
}

__global__ void vadd_f32(const float* __restrict__ a, const float* __restrict__ b,
                         float* __restrict__ c, int n) {
  int i = blockIdx.x * blockDim.x + threadIdx.x;
  if (i < n) c[i] = a[i] + b[i];
}

// x_f16[(t*16+b), e] = embed[tokens[b*T+t], e]   (padding row 0 already zero)
__global__ void embed_gather(const int* __restrict__ tok, const float* __restrict__ emb,
                             _Float16* __restrict__ xh) {
  int tb = blockIdx.x;              // t*16 + b
  int t = tb >> 4, b = tb & 15;
  int token = tok[b * 256 + t];
  const float* src = emb + (long)token * 512;
  _Float16* dst = xh + (long)tb * 512;
  for (int e = threadIdx.x; e < 512; e += blockDim.x) dst[e] = (_Float16)src[e];
}

__global__ void init_state(_Float16* __restrict__ h0g, _Float16* __restrict__ h1g,
                           unsigned* __restrict__ barcnt) {
  int i = blockIdx.x * blockDim.x + threadIdx.x;
  const int n = 2 * 16 * 512;
  if (i < n) { h0g[i] = (_Float16)0.0f; h1g[i] = (_Float16)0.0f; }
  if (i == 0) *barcnt = 0u;
}

// ------------------------------- generic WMMA GEMM -------------------------
// C[m,n] = sum_k A[m,k] * W[n,k] + bias[n];  A:[M,K] f16, W:[N,K] f16,
// C:[M,N] f32.  One wave -> 32(M) x 64(N) tile: 2 A-frags x 4 B-frags -> 8
// WMMAs per k-step.  Ping-pong fragment sets (unroll-2) for latency hiding.
// Requires K a multiple of 64 (K = 512 here).

struct FragSet { v16h a0, a1, b0, b1, b2, b3; };

__device__ __forceinline__ void load_set(FragSet& s, const _Float16* At0,
                                         const _Float16* At1, const _Float16* wr,
                                         int K, int kb, int r, int hi) {
  s.a0 = load_a_frag(At0, K, kb, r, hi);
  s.a1 = load_a_frag(At1, K, kb, r, hi);
  s.b0 = load_b_frag(wr, kb, hi);
  s.b1 = load_b_frag(wr + (long)16 * K, kb, hi);
  s.b2 = load_b_frag(wr + (long)32 * K, kb, hi);
  s.b3 = load_b_frag(wr + (long)48 * K, kb, hi);
}

__device__ __forceinline__ void compute_set(const FragSet& s, v8f acc[2][4]) {
  acc[0][0] = WMMA16(s.a0, s.b0, acc[0][0]);
  acc[1][0] = WMMA16(s.a1, s.b0, acc[1][0]);
  acc[0][1] = WMMA16(s.a0, s.b1, acc[0][1]);
  acc[1][1] = WMMA16(s.a1, s.b1, acc[1][1]);
  acc[0][2] = WMMA16(s.a0, s.b2, acc[0][2]);
  acc[1][2] = WMMA16(s.a1, s.b2, acc[1][2]);
  acc[0][3] = WMMA16(s.a0, s.b3, acc[0][3]);
  acc[1][3] = WMMA16(s.a1, s.b3, acc[1][3]);
}

__global__ __launch_bounds__(256)
void gemm_bias_wmma(const _Float16* __restrict__ A, const _Float16* __restrict__ W,
                    const float* __restrict__ bias, float* __restrict__ C,
                    int M, int N, int K) {
  int gwave = (int)((blockIdx.x * blockDim.x + threadIdx.x) >> 5);
  int lane = threadIdx.x & 31;
  int r = lane & 15, hi = lane >> 4;
  int nt64 = N >> 6;
  int nwaves = (M >> 5) * nt64;
  if (gwave >= nwaves) return;
  int mt = gwave / nt64, nt = gwave % nt64;
  int row0 = mt << 5, n0 = nt << 6;

  const _Float16* At0 = A + (long)row0 * K;
  const _Float16* At1 = At0 + (long)16 * K;
  const _Float16* wr = W + (long)(n0 + r) * K;   // this lane's weight rows

  v8f acc[2][4];
#pragma unroll
  for (int j = 0; j < 4; ++j) {
    float bv = bias[n0 + j * 16 + r];
#pragma unroll
    for (int i = 0; i < 8; ++i) { acc[0][j][i] = bv; acc[1][j][i] = bv; }
  }

  FragSet s0, s1;
  load_set(s0, At0, At1, wr, K, 0, r, hi);
  int kb = 0;
  for (; kb + 64 < K; kb += 64) {
    load_set(s1, At0, At1, wr, K, kb + 32, r, hi);
    compute_set(s0, acc);
    load_set(s0, At0, At1, wr, K, kb + 64, r, hi);
    compute_set(s1, acc);
  }
  load_set(s1, At0, At1, wr, K, kb + 32, r, hi);   // tail pair
  compute_set(s0, acc);
  compute_set(s1, acc);

#pragma unroll
  for (int h = 0; h < 2; ++h) {
#pragma unroll
    for (int j = 0; j < 4; ++j) {
#pragma unroll
      for (int i = 0; i < 8; ++i) {
        int m = row0 + h * 16 + i + 8 * hi;
        C[(long)m * N + n0 + j * 16 + r] = acc[h][j][i];
      }
    }
  }
}

// ------------------------------- sequential LSTM ---------------------------
// 32 workgroups x 256 threads. WG w owns gate columns [w*16, w*16+16) of each
// of the 4 gates for both layers; weight slices resident in LDS for all 256
// steps.  Wave = (khalf<<2)|gate : one 16x16 WMMA tile over half of K.
// h0/h1 double-buffered in global memory; device-scope atomic counter barrier.

__device__ __forceinline__ void gbarrier(unsigned* cnt, unsigned target) {
  __threadfence();
  __syncthreads();
  if (threadIdx.x == 0) {
    __hip_atomic_fetch_add(cnt, 1u, __ATOMIC_RELEASE, __HIP_MEMORY_SCOPE_AGENT);
    while (__hip_atomic_load(cnt, __ATOMIC_ACQUIRE, __HIP_MEMORY_SCOPE_AGENT) < target)
      __builtin_amdgcn_s_sleep(4);
  }
  __syncthreads();
}

__global__ __launch_bounds__(256)
void lstm_seq(const float* __restrict__ preact0,    // [T*16, 2048]
              const _Float16* __restrict__ Wh0,     // [2048, 512]
              const _Float16* __restrict__ Wi1,     // [2048, 512]
              const _Float16* __restrict__ Wh1,     // [2048, 512]
              const float* __restrict__ bsum1,      // [2048] = bi1+bh1
              _Float16* __restrict__ h0g,           // [2][16][512]
              _Float16* __restrict__ h1g,           // [2][16][512]
              unsigned* __restrict__ barcnt,
              _Float16* __restrict__ feats,         // [B*T, 512], row b*T+t
              int nblocks, int T) {
  const int WKP = 520;  // padded LDS row stride (f16) -> conflict-free B frags
  extern __shared__ char smem[];
  _Float16* ws0 = (_Float16*)smem;                   // Wh0 slice [64][520]
  _Float16* wsI = ws0 + 64 * WKP;                    // Wi1 slice
  _Float16* wsH = wsI + 64 * WKP;                    // Wh1 slice
  float* accs = (float*)(wsH + 64 * WKP);            // [8][16][16] partial D
  float* c0s = accs + 8 * 16 * 16;                   // [16][16] cell state L0
  float* c1s = c0s + 16 * 16;                        // [16][16] cell state L1

  int w = blockIdx.x;                // 0..31 : gate-column slice
  int tid = threadIdx.x;
  int wave = tid >> 5, lane = tid & 31;
  int r = lane & 15, hi = lane >> 4;
  int g = wave & 3, kh = wave >> 2;  // gate id, K half

  // Stage weight slices into LDS (rows: gate*16 + r, full K=512).
  for (int c = tid; c < 64 * 64; c += 256) {   // 64 chunks of 8 f16 per row
    int row = c >> 6;
    int off = (c & 63) * 8;
    int gg = row >> 4, rr = row & 15;
    long grow = (long)(gg * 512 + w * 16 + rr) * 512 + off;
    *(v8h*)(ws0 + row * WKP + off) = *(const v8h*)(Wh0 + grow);
    *(v8h*)(wsI + row * WKP + off) = *(const v8h*)(Wi1 + grow);
    *(v8h*)(wsH + row * WKP + off) = *(const v8h*)(Wh1 + grow);
  }
  for (int c = tid; c < 16 * 16; c += 256) { c0s[c] = 0.0f; c1s[c] = 0.0f; }
  __syncthreads();

  unsigned phase = 0;
  int kbase = kh * 256;
  const _Float16* wrow0 = ws0 + (g * 16 + r) * WKP + kbase;
  const _Float16* wrowI = wsI + (g * 16 + r) * WKP + kbase;
  const _Float16* wrowH = wsH + (g * 16 + r) * WKP + kbase;

  for (int t = 0; t < T; ++t) {
    int cur = t & 1, prev = cur ^ 1;

    // ---- layer 0 recurrent GEMM: partial over K half kh ----
    {
      v8f acc = vzero8();
      const _Float16* hprev = h0g + prev * 16 * 512;
      v16h a = load_a_frag(hprev, 512, kbase, r, hi);
      v16h b = load_b_frag(wrow0, 0, hi);
#pragma unroll
      for (int kt = 0; kt < 8; ++kt) {
        v16h na = a, nb = b;
        if (kt < 7) {
          na = load_a_frag(hprev, 512, kbase + (kt + 1) * 32, r, hi);
          nb = load_b_frag(wrow0, (kt + 1) * 32, hi);
        }
        acc = WMMA16(a, b, acc);
        a = na; b = nb;
      }
#pragma unroll
      for (int i = 0; i < 8; ++i) accs[(wave * 16 + i + 8 * hi) * 16 + r] = acc[i];
    }
    __syncthreads();
    // gate math: one thread per (m, col) cell of this WG's 16 columns
    {
      int m = tid >> 4, col = tid & 15;
      const float* pr = preact0 + (long)(t * 16 + m) * 2048 + w * 16 + col;
      float iv = accs[(0 * 16 + m) * 16 + col] + accs[(4 * 16 + m) * 16 + col] + pr[0];
      float fv = accs[(1 * 16 + m) * 16 + col] + accs[(5 * 16 + m) * 16 + col] + pr[512];
      float gv = accs[(2 * 16 + m) * 16 + col] + accs[(6 * 16 + m) * 16 + col] + pr[1024];
      float ov = accs[(3 * 16 + m) * 16 + col] + accs[(7 * 16 + m) * 16 + col] + pr[1536];
      float cc = sigf(fv) * c0s[m * 16 + col] + sigf(iv) * tanhf(gv);
      c0s[m * 16 + col] = cc;
      float hh = sigf(ov) * tanhf(cc);
      h0g[cur * 16 * 512 + m * 512 + w * 16 + col] = (_Float16)hh;
    }
    gbarrier(barcnt, (++phase) * (unsigned)nblocks);   // h0_t visible everywhere

    // ---- layer 1 GEMM: [h0_t | h1_{t-1}] x [Wi1 | Wh1], partial K half ----
    {
      v8f acc = vzero8();
      const _Float16* h0c = h0g + cur * 16 * 512;
      const _Float16* h1p = h1g + prev * 16 * 512;
      v16h a = load_a_frag(h0c, 512, kbase, r, hi);
      v16h b = load_b_frag(wrowI, 0, hi);
#pragma unroll
      for (int kt = 0; kt < 8; ++kt) {       // Wi1 half; preload crosses into Wh1
        v16h na, nb;
        if (kt < 7) {
          na = load_a_frag(h0c, 512, kbase + (kt + 1) * 32, r, hi);
          nb = load_b_frag(wrowI, (kt + 1) * 32, hi);
        } else {
          na = load_a_frag(h1p, 512, kbase, r, hi);
          nb = load_b_frag(wrowH, 0, hi);
        }
        acc = WMMA16(a, b, acc);
        a = na; b = nb;
      }
#pragma unroll
      for (int kt = 0; kt < 8; ++kt) {       // Wh1 half
        v16h na = a, nb = b;
        if (kt < 7) {
          na = load_a_frag(h1p, 512, kbase + (kt + 1) * 32, r, hi);
          nb = load_b_frag(wrowH, (kt + 1) * 32, hi);
        }
        acc = WMMA16(a, b, acc);
        a = na; b = nb;
      }
#pragma unroll
      for (int i = 0; i < 8; ++i) accs[(wave * 16 + i + 8 * hi) * 16 + r] = acc[i];
    }
    __syncthreads();
    {
      int m = tid >> 4, col = tid & 15;
      int n = w * 16 + col;
      float iv = accs[(0 * 16 + m) * 16 + col] + accs[(4 * 16 + m) * 16 + col] + bsum1[n];
      float fv = accs[(1 * 16 + m) * 16 + col] + accs[(5 * 16 + m) * 16 + col] + bsum1[512 + n];
      float gv = accs[(2 * 16 + m) * 16 + col] + accs[(6 * 16 + m) * 16 + col] + bsum1[1024 + n];
      float ov = accs[(3 * 16 + m) * 16 + col] + accs[(7 * 16 + m) * 16 + col] + bsum1[1536 + n];
      float cc = sigf(fv) * c1s[m * 16 + col] + sigf(iv) * tanhf(gv);
      c1s[m * 16 + col] = cc;
      float hh = sigf(ov) * tanhf(cc);
      h1g[cur * 16 * 512 + m * 512 + n] = (_Float16)hh;
      feats[((long)m * T + t) * 512 + n] = (_Float16)hh;   // row b*T+t
    }
    gbarrier(barcnt, (++phase) * (unsigned)nblocks);   // h1_t visible everywhere
  }
}

// ------------------------------- launcher ----------------------------------

extern "C" void kernel_launch(void* const* d_in, const int* in_sizes, int n_in,
                              void* d_out, int out_size, void* d_ws, size_t ws_size,
                              hipStream_t stream) {
  (void)in_sizes; (void)n_in; (void)out_size; (void)ws_size;
  const int*   tokens = (const int*)d_in[2];
  const float* embed  = (const float*)d_in[3];
  const float* Wi0    = (const float*)d_in[4];
  const float* bi0    = (const float*)d_in[5];
  const float* Wh0    = (const float*)d_in[6];
  const float* bh0    = (const float*)d_in[7];
  const float* Wi1    = (const float*)d_in[8];
  const float* bi1    = (const float*)d_in[9];
  const float* Wh1    = (const float*)d_in[10];
  const float* bh1    = (const float*)d_in[11];
  const float* Wout   = (const float*)d_in[12];
  const float* bout   = (const float*)d_in[13];
  float* out = (float*)d_out;

  char* ws = (char*)d_ws;
  size_t off = 0;
  auto carve = [&](size_t bytes) -> char* {
    char* p = ws + off;
    off = (off + bytes + 255) & ~(size_t)255;
    return p;
  };
  _Float16* xh      = (_Float16*)carve((size_t)4096 * 512 * 2);
  _Float16* Wi0h    = (_Float16*)carve((size_t)2048 * 512 * 2);
  _Float16* Wh0h    = (_Float16*)carve((size_t)2048 * 512 * 2);
  _Float16* Wi1h    = (_Float16*)carve((size_t)2048 * 512 * 2);
  _Float16* Wh1h    = (_Float16*)carve((size_t)2048 * 512 * 2);
  _Float16* Wouth   = (_Float16*)carve((size_t)32000 * 512 * 2);
  float*    preact0 = (float*)carve((size_t)4096 * 2048 * 4);
  _Float16* feats   = (_Float16*)carve((size_t)4096 * 512 * 2);
  float*    bsum0   = (float*)carve((size_t)2048 * 4);
  float*    bsum1   = (float*)carve((size_t)2048 * 4);
  _Float16* h0g     = (_Float16*)carve((size_t)2 * 16 * 512 * 2);
  _Float16* h1g     = (_Float16*)carve((size_t)2 * 16 * 512 * 2);
  unsigned* barcnt  = (unsigned*)carve(256);

  // 1) weight conversion + bias sums
  cvt_f16<<<1024, 256, 0, stream>>>(Wi0, Wi0h, 2048L * 512);
  cvt_f16<<<1024, 256, 0, stream>>>(Wh0, Wh0h, 2048L * 512);
  cvt_f16<<<1024, 256, 0, stream>>>(Wi1, Wi1h, 2048L * 512);
  cvt_f16<<<1024, 256, 0, stream>>>(Wh1, Wh1h, 2048L * 512);
  cvt_f16<<<4096, 256, 0, stream>>>(Wout, Wouth, 32000L * 512);
  vadd_f32<<<8, 256, 0, stream>>>(bi0, bh0, bsum0, 2048);
  vadd_f32<<<8, 256, 0, stream>>>(bi1, bh1, bsum1, 2048);

  // 2) embedding gather (rows t*16+b)
  embed_gather<<<4096, 256, 0, stream>>>(tokens, embed, xh);

  // 3) preact0 = x @ Wi0^T + (bi0+bh0) : M=4096, N=2048, K=512
  //    waves = (4096/32)*(2048/64) = 4096 -> 512 blocks of 8 waves
  gemm_bias_wmma<<<512, 256, 0, stream>>>(xh, Wi0h, bsum0, preact0, 4096, 2048, 512);

  // 4) sequential LSTM: 32 WGs, 195KB LDS weight residency + staging
  init_state<<<64, 256, 0, stream>>>(h0g, h1g, barcnt);
  size_t smem = (size_t)3 * 64 * 520 * 2 + (size_t)8 * 16 * 16 * 4 + (size_t)2 * 16 * 16 * 4;
  lstm_seq<<<32, 256, smem, stream>>>(preact0, Wh0h, Wi1h, Wh1h, bsum1,
                                      h0g, h1g, barcnt, feats, 32, 256);

  // 5) out = feats @ Wout^T + bout : M=4096, N=32000, K=512
  //    waves = (4096/32)*(32000/64) = 64000 -> 8000 blocks of 8 waves
  gemm_bias_wmma<<<8000, 256, 0, stream>>>(feats, Wouth, bout, out, 4096, 32000, 512);
}
